// flash_attn1_76579266888121
// MI455X (gfx1250) — compile-verified, matrix operand order checked by reading, not run on hardware
//
#include <hip/hip_runtime.h>
#include <hip/hip_bf16.h>

typedef __attribute__((ext_vector_type(16))) __bf16 v16bf;
typedef __attribute__((ext_vector_type(2)))  __bf16 v2bf;
typedef __attribute__((ext_vector_type(2)))  float  v2f;
typedef __attribute__((ext_vector_type(8)))  float  v8f;

#define S_LEN 8192
#define D_DIM 1024
#define BM 64            // query rows per workgroup
#define BN 128           // keys per KV iteration
#define QS 1028          // Q LDS row stride (bf16, padded; 514 dw = 2 mod 64)
#define KSS 68           // K chunk row stride (bf16): 64 feats + 4 pad
#define VSS 132          // Vt row stride (bf16): 128 keys + 4 pad
#define SBS 130          // S buffer row stride (f32)
#define PBS 132          // P buffer row stride (bf16)

// Scalar f32 -> bf16 (RNE): lowers to v_cvt_pk_bf16_f32 low-half.
__device__ __forceinline__ unsigned short f2bf(float f) {
  union { __bf16 h; unsigned short u; } c;
  c.h = (__bf16)f;
  return c.u;
}
// Packed pair conversion: single v_cvt_pk_bf16_f32 vD, vA, vB.
__device__ __forceinline__ unsigned pk2bf(float a, float b) {
  union { v2bf h; unsigned u; } c;
  c.h = __builtin_convertvector((v2f){a, b}, v2bf);
  return c.u;
}

union FragU { v16bf v; unsigned u[8]; };

// ISA 7.12.2 16-bit A/B fragment: lane m holds row m (both halves), dword j
// holds the bf16 pair at k = 2*(j&3) + 16*(j>>2) + 8*(lane>=16).
__device__ __forceinline__ v16bf load_frag(const unsigned short* base, int stride, int lane) {
  const int m = lane & 15, h = lane >> 4;
  const unsigned short* p = base + m * stride + 8 * h;
  FragU f;
#pragma unroll
  for (int j = 0; j < 8; ++j) {
    const int k = 2 * (j & 3) + 16 * (j >> 2);
    f.u[j] = *(const unsigned*)(p + k);
  }
  return f.v;
}

#define WMMA_BF16(A, B, C) \
  __builtin_amdgcn_wmma_f32_16x16x32_bf16(false, (A), false, (B), (short)0, (C), false, false)

__global__ __launch_bounds__(256) void fa4_cdna5_kernel(
    const float* __restrict__ Q, const float* __restrict__ K,
    const float* __restrict__ V, float* __restrict__ O) {
  __shared__ unsigned short Qs[BM * QS];      // Q tile bf16            (131.6 KB)
  __shared__ unsigned short Ks[BN * KSS];     // K chunk: 128 keys x 64 feats (17.4 KB)
  __shared__ unsigned short Vt[64 * VSS];     // V chunk transposed: 64 cols x 128 keys (16.9 KB)
  __shared__ float          Sb[BM * SBS];     // S tile f32             (33.3 KB)
  __shared__ unsigned short Pb[BM * PBS];     // P tile bf16            (16.9 KB)
  __shared__ float pmx[BM * 4], psm[BM * 4];  // softmax partials
  __shared__ float mb[BM], lb[BM], ab[BM], mnb[BM];

  const int tid  = threadIdx.x;
  const int lane = tid & 31;
  const int wave = tid >> 5;
  const int rw32 = (wave >> 2) * 32;   // row-group base: 0 or 32
  const int cw   = wave & 3;           // col group 0..3
  const int hl   = lane >> 4, nl = lane & 15;
  const int qbase = blockIdx.x * BM;

  // ---- stage Q (f32 -> bf16, packed dword stores) ----
  {
    const float4* Q4 = (const float4*)(Q + (size_t)qbase * D_DIM);
#pragma unroll
    for (int i = 0; i < 64; ++i) {
      const int e4 = tid + 256 * i;              // 16384 float4
      const float4 q = Q4[e4];
      const int row = e4 >> 8, c4 = e4 & 255;
      unsigned* dst = (unsigned*)&Qs[row * QS + 4 * c4];
      dst[0] = pk2bf(q.x, q.y);
      dst[1] = pk2bf(q.z, q.w);
    }
  }
  if (tid < BM) { mb[tid] = -__builtin_inff(); lb[tid] = 0.0f; }

  v8f acc[32];                          // 2 row-tiles x 16 col-tiles = 256 VGPRs
#pragma unroll
  for (int t = 0; t < 32; ++t) acc[t] = v8f{0,0,0,0,0,0,0,0};

  for (int j = 0; j < S_LEN / BN; ++j) {        // 64 KV iterations
    const int kb = j * BN;
    v8f sacc[4];
#pragma unroll
    for (int t = 0; t < 4; ++t) sacc[t] = v8f{0,0,0,0,0,0,0,0};

    // ---- S = Q K^T : D in chunks of 64, 2x2 WMMA block per wave ----
#pragma unroll 1
    for (int dc = 0; dc < 16; ++dc) {
      const int db = dc * 64;
      if (dc < 15) {  // prefetch next K chunk (global_prefetch_b8)
        const float* nk = K + (size_t)kb * D_DIM + (size_t)(tid >> 1) * D_DIM
                            + (db + 64) + (tid & 1) * 32;
        __builtin_prefetch(nk, 0, 0);
      }
      __syncthreads();
      {
        const float4* K4 = (const float4*)(K + (size_t)kb * D_DIM + db);
#pragma unroll
        for (int i = 0; i < 8; ++i) {
          const int e4 = tid + 256 * i;          // 128 rows x 16 float4
          const int r = e4 >> 4, c4 = e4 & 15;
          const float4 k = K4[(size_t)r * (D_DIM / 4) + c4];
          unsigned* dst = (unsigned*)&Ks[r * KSS + 4 * c4];
          dst[0] = pk2bf(k.x, k.y);
          dst[1] = pk2bf(k.z, k.w);
        }
      }
      __syncthreads();
#pragma unroll
      for (int ks = 0; ks < 2; ++ks) {
        const int kk = 32 * ks;
        v16bf a0 = load_frag(&Qs[(rw32 +  0) * QS + db + kk], QS, lane);
        v16bf a1 = load_frag(&Qs[(rw32 + 16) * QS + db + kk], QS, lane);
        v16bf b0 = load_frag(&Ks[(32 * cw +  0) * KSS + kk], KSS, lane);
        v16bf b1 = load_frag(&Ks[(32 * cw + 16) * KSS + kk], KSS, lane);
        sacc[0] = WMMA_BF16(a0, b0, sacc[0]);
        sacc[1] = WMMA_BF16(a0, b1, sacc[1]);
        sacc[2] = WMMA_BF16(a1, b0, sacc[2]);
        sacc[3] = WMMA_BF16(a1, b1, sacc[3]);
      }
    }
    // scatter S tiles (C layout: row = v + 8*hl, col = nl)
#pragma unroll
    for (int rt = 0; rt < 2; ++rt)
#pragma unroll
      for (int ct = 0; ct < 2; ++ct)
#pragma unroll
        for (int v = 0; v < 8; ++v)
          Sb[(rw32 + 16 * rt + v + 8 * hl) * SBS + 32 * cw + 16 * ct + nl] = sacc[rt * 2 + ct][v];
    __syncthreads();

    // ---- parallel online softmax: 4 threads per row, 32 cols each ----
    {
      const int r = tid >> 2, sub = tid & 3;
      float pm = -__builtin_inff();
      for (int c = 0; c < 32; ++c) pm = fmaxf(pm, Sb[r * SBS + 32 * sub + c]);
      pmx[r * 4 + sub] = pm;
    }
    __syncthreads();
    if ((tid & 3) == 0) {
      const int r = tid >> 2;
      const float mij = fmaxf(fmaxf(pmx[4 * r], pmx[4 * r + 1]),
                              fmaxf(pmx[4 * r + 2], pmx[4 * r + 3]));
      const float mo = mb[r], mn = fmaxf(mo, mij);
      ab[r] = __expf(mo - mn);           // 0 on first step
      mb[r] = mn; mnb[r] = mn;
    }
    __syncthreads();
    {
      const int r = tid >> 2, sub = tid & 3;
      const float mn = mnb[r];
      float sum = 0.0f;
      for (int c = 0; c < 16; ++c) {     // 16 pair-steps = 32 cols
        const float p0 = __expf(Sb[r * SBS + 32 * sub + 2 * c]     - mn);
        const float p1 = __expf(Sb[r * SBS + 32 * sub + 2 * c + 1] - mn);
        sum += p0 + p1;
        *(unsigned*)&Pb[r * PBS + 32 * sub + 2 * c] = pk2bf(p0, p1);
      }
      psm[r * 4 + sub] = sum;
    }
    __syncthreads();
    if ((tid & 3) == 0) {
      const int r = tid >> 2;
      lb[r] = ab[r] * lb[r] + psm[4 * r] + psm[4 * r + 1] + psm[4 * r + 2] + psm[4 * r + 3];
    }
    __syncthreads();

    // ---- rescale O, cache P fragments in registers ----
    float al[2][8];
#pragma unroll
    for (int rt = 0; rt < 2; ++rt)
#pragma unroll
      for (int v = 0; v < 8; ++v) al[rt][v] = ab[rw32 + 16 * rt + v + 8 * hl];
    v16bf pf[2][4];
#pragma unroll
    for (int rt = 0; rt < 2; ++rt)
#pragma unroll
      for (int kc = 0; kc < 4; ++kc)
        pf[rt][kc] = load_frag(&Pb[(rw32 + 16 * rt) * PBS + 32 * kc], PBS, lane);
#pragma unroll
    for (int t = 0; t < 32; ++t)
#pragma unroll
      for (int v = 0; v < 8; ++v) acc[t][v] *= al[t >> 4][v];

    // ---- O += P V : D in chunks of 64 (V transposed), fully unrolled ----
#pragma unroll
    for (int cc = 0; cc < 16; ++cc) {
      const int cb = 64 * cc;
      __syncthreads();
      {
        const float4* V4 = (const float4*)(V + (size_t)kb * D_DIM + cb);
#pragma unroll
        for (int i = 0; i < 8; ++i) {
          const int e4 = tid + 256 * i;
          const int r = e4 >> 4, c4 = e4 & 15;
          const float4 v = V4[(size_t)r * (D_DIM / 4) + c4];
          Vt[(4 * c4 + 0) * VSS + r] = f2bf(v.x);   // transpose: scalar b16 stores
          Vt[(4 * c4 + 1) * VSS + r] = f2bf(v.y);
          Vt[(4 * c4 + 2) * VSS + r] = f2bf(v.z);
          Vt[(4 * c4 + 3) * VSS + r] = f2bf(v.w);
        }
      }
      __syncthreads();
#pragma unroll
      for (int kc = 0; kc < 4; ++kc) {
        v16bf bv = load_frag(&Vt[(16 * cw) * VSS + 32 * kc], VSS, lane);
        acc[cc]      = WMMA_BF16(pf[0][kc], bv, acc[cc]);
        acc[16 + cc] = WMMA_BF16(pf[1][kc], bv, acc[16 + cc]);
      }
    }
  }
  __syncthreads();

  // ---- finalize: O /= l ----
  float li[2][8];
#pragma unroll
  for (int rt = 0; rt < 2; ++rt)
#pragma unroll
    for (int v = 0; v < 8; ++v) li[rt][v] = 1.0f / lb[rw32 + 16 * rt + v + 8 * hl];
#pragma unroll
  for (int rt = 0; rt < 2; ++rt)
#pragma unroll
    for (int ct = 0; ct < 16; ++ct) {
      const int col = 64 * ct + 16 * cw + nl;    // global col tile = 4*ct + cw
#pragma unroll
      for (int v = 0; v < 8; ++v) {
        const int row = qbase + rw32 + 16 * rt + v + 8 * hl;
        O[(size_t)row * D_DIM + col] = acc[rt * 16 + ct][v] * li[rt][v];
      }
    }
}

extern "C" void kernel_launch(void* const* d_in, const int* in_sizes, int n_in,
                              void* d_out, int out_size, void* d_ws, size_t ws_size,
                              hipStream_t stream) {
  const float* Q = (const float*)d_in[0];
  const float* K = (const float*)d_in[1];
  const float* V = (const float*)d_in[2];
  float* O = (float*)d_out;
  dim3 grid(S_LEN / BM), block(256);
  hipLaunchKernelGGL(fa4_cdna5_kernel, grid, block, 0, stream, Q, K, V, O);
}
